// SoftPatternClassifier_3504693313664
// MI455X (gfx1250) — compile-verified, matrix operand order checked by reading, not run on hardware
//
#include <hip/hip_runtime.h>

// ---------------------------------------------------------------------------
// SoPa (soft patterns) classifier, fused for MI455X (gfx1250, wave32).
//
//   1) prep_b : diags fp32 -> bf16 hi/lo, pre-swizzled WMMA B layout (2.7 MB)
//   2) prep_a : gather embedding columns per token, bf16 hi/lo WMMA A layout
//               (84 MB; embedding matrix is 60 MB -> L2-resident gathers)
//   3) sopa_main : per (batch, 8-pattern block): 7 GEMM waves hold their B
//               panel resident in VGPRs and compute 16x112 fp32 transition
//               tiles via bf16x3 WMMA; A tiles are staged into LDS with
//               GLOBAL_LOAD_ASYNC_TO_LDS_B128 one tile ahead (double
//               buffered); A operands are software-pipelined one K-step
//               ahead out of LDS; 1 scan wave consumes tiles concurrently.
//   4) mlp    : 150 -> 100 -> relu -> 2 head.
// ---------------------------------------------------------------------------

typedef __attribute__((ext_vector_type(16))) __bf16    v16bf;
typedef __attribute__((ext_vector_type(8)))  float     v8f;
typedef __attribute__((ext_vector_type(4)))  unsigned  uint4v;
typedef __attribute__((ext_vector_type(4)))  int       v4i;

#define SP_B   32
#define SP_T   2048
#define SP_V   50000
#define SP_WD  300
#define SP_P   150
#define SP_COLS      2100
#define SP_KSTEPS    10        // K padded 300 -> 320
#define SP_CTILES    133
#define SP_TTILES    128
#define SP_PBLOCKS   19
#define SP_ZERO      (-100.0f)

#if defined(__has_builtin)
#if __has_builtin(__builtin_amdgcn_global_load_async_to_lds_b128)
#define SP_HAVE_ASYNC 1
#endif
#endif
#ifndef SP_HAVE_ASYNC
#define SP_HAVE_ASYNC 0
#endif

// Copy one 32-byte per-lane chunk global -> LDS (async, ASYNCcnt-tracked).
__device__ __forceinline__ void copy32_to_lds(const unsigned* g, unsigned* l) {
#if SP_HAVE_ASYNC
  v4i* gp = (v4i*)(uintptr_t)g;
  v4i* lp = (v4i*)(uintptr_t)l;
  __builtin_amdgcn_global_load_async_to_lds_b128(gp, lp, 0, 0);
  __builtin_amdgcn_global_load_async_to_lds_b128(gp + 1, lp + 1, 0, 0);
#else
  *(uint4v*)l       = *(const uint4v*)g;
  *((uint4v*)l + 1) = *((const uint4v*)g + 1);
#endif
}

__device__ __forceinline__ void wait_async_copies() {
#if SP_HAVE_ASYNC
#if defined(__has_builtin) && __has_builtin(__builtin_amdgcn_s_wait_asynccnt)
  __builtin_amdgcn_s_wait_asynccnt(0);
#else
  asm volatile("s_wait_asynccnt 0x0" ::: "memory");
#endif
#endif
}

// round-to-nearest-even fp32 -> bf16
__device__ __forceinline__ unsigned short f2bf(float x) {
  unsigned u = __float_as_uint(x);
  unsigned r = u + 0x7FFFu + ((u >> 16) & 1u);
  return (unsigned short)(r >> 16);
}
__device__ __forceinline__ float bf2f(unsigned short h) {
  return __uint_as_float(((unsigned)h) << 16);
}

// K position inside a 16-bit WMMA operand for (dword v, lane-half, 16b half)
__device__ __forceinline__ int klocal_of(int v, int laneHi, int half) {
  int base = (v < 4) ? (2 * v) : (16 + 2 * (v - 4));
  return base + laneHi * 8 + half;
}

// ---------------------------------------------------------------------------
// prep_b: grid = 133, block = 320.  Bq[((ct*10+ks)*32+lane)*8 + v]
// ---------------------------------------------------------------------------
__global__ void __launch_bounds__(320) prep_b_kernel(
    const float* __restrict__ diags, unsigned* __restrict__ Bh,
    unsigned* __restrict__ Bl) {
  const int ct = blockIdx.x, ks = threadIdx.x >> 5, lane = threadIdx.x & 31;
  const int laneHi = lane >> 4;
  const int j = ct * 16 + (lane & 15);

  unsigned oh[8], ol[8];
#pragma unroll
  for (int v = 0; v < 8; ++v) {
    unsigned short hh[2], ll[2];
#pragma unroll
    for (int half = 0; half < 2; ++half) {
      int k = ks * 32 + klocal_of(v, laneHi, half);
      float f = (j < SP_COLS && k < SP_WD) ? diags[(size_t)j * SP_WD + k] : 0.0f;
      unsigned short h = f2bf(f);
      hh[half] = h;
      ll[half] = f2bf(f - bf2f(h));
    }
    oh[v] = (unsigned)hh[0] | ((unsigned)hh[1] << 16);
    ol[v] = (unsigned)ll[0] | ((unsigned)ll[1] << 16);
  }
  size_t base = ((size_t)(ct * SP_KSTEPS + ks) * 32 + lane) * 8;
  *(uint4v*)(Bh + base)     = (uint4v){oh[0], oh[1], oh[2], oh[3]};
  *(uint4v*)(Bh + base + 4) = (uint4v){oh[4], oh[5], oh[6], oh[7]};
  *(uint4v*)(Bl + base)     = (uint4v){ol[0], ol[1], ol[2], ol[3]};
  *(uint4v*)(Bl + base + 4) = (uint4v){ol[4], ol[5], ol[6], ol[7]};
}

// ---------------------------------------------------------------------------
// prep_a: grid = (128, 32), block = 320.
// Aq[(((b*128+tt)*10+ks)*32+lane)*8 + v]
// ---------------------------------------------------------------------------
__global__ void __launch_bounds__(320) prep_a_kernel(
    const float* __restrict__ emb, const int* __restrict__ docs,
    unsigned* __restrict__ Ah, unsigned* __restrict__ Al) {
  const int tt = blockIdx.x, b = blockIdx.y;
  const int ks = threadIdx.x >> 5, lane = threadIdx.x & 31;
  const int laneHi = lane >> 4;
  const int t = tt * 16 + (lane & 15);
  const int tok = docs[b * SP_T + t];

  unsigned oh[8], ol[8];
#pragma unroll
  for (int v = 0; v < 8; ++v) {
    unsigned short hh[2], ll[2];
#pragma unroll
    for (int half = 0; half < 2; ++half) {
      int k = ks * 32 + klocal_of(v, laneHi, half);
      float f = (k < SP_WD) ? emb[(size_t)k * SP_V + tok] : 0.0f;
      unsigned short h = f2bf(f);
      hh[half] = h;
      ll[half] = f2bf(f - bf2f(h));
    }
    oh[v] = (unsigned)hh[0] | ((unsigned)hh[1] << 16);
    ol[v] = (unsigned)ll[0] | ((unsigned)ll[1] << 16);
  }
  size_t base = (((size_t)(b * SP_TTILES + tt) * SP_KSTEPS + ks) * 32 + lane) * 8;
  *(uint4v*)(Ah + base)     = (uint4v){oh[0], oh[1], oh[2], oh[3]};
  *(uint4v*)(Ah + base + 4) = (uint4v){oh[4], oh[5], oh[6], oh[7]};
  *(uint4v*)(Al + base)     = (uint4v){ol[0], ol[1], ol[2], ol[3]};
  *(uint4v*)(Al + base + 4) = (uint4v){ol[4], ol[5], ol[6], ol[7]};
}

// ---------------------------------------------------------------------------
// Main fused kernel.  grid = (19, 32), block = 256 (8 waves).
// Waves 0..6: GEMM; wave 7: max-plus scan (lanes 0..7 = patterns).
// ---------------------------------------------------------------------------
__global__ void __launch_bounds__(256) sopa_main_kernel(
    const unsigned* __restrict__ Ahq, const unsigned* __restrict__ Alq,
    const unsigned* __restrict__ Bhq, const unsigned* __restrict__ Blq,
    const float* __restrict__ bias, const float* __restrict__ epsilon,
    const int* __restrict__ doc_lens, float* __restrict__ scores) {
  __shared__ float tmLDS[2][16][112];                                 // 14 KB
  __shared__ __attribute__((aligned(32)))
      unsigned aLDS[2][2][SP_KSTEPS][32][8];                          // 40 KB

  const int pblock = blockIdx.x;        // 0..18
  const int b      = blockIdx.y;        // 0..31
  const int w      = threadIdx.x >> 5;  // 0..7
  const int lane   = threadIdx.x & 31;
  const bool gemmWave = (w < 7);

  // ---- GEMM-wave persistent state: resident B panel ----
  v16bf Bh[SP_KSTEPS], Bl[SP_KSTEPS];
  float bv = 0.0f;
  int colLocal = 0, rowBase = 0;
  size_t abase = 0;
  if (gemmWave) {
    const int gct = pblock * 7 + w;
#pragma unroll
    for (int ks = 0; ks < SP_KSTEPS; ++ks) {
      size_t off = ((size_t)(gct * SP_KSTEPS + ks) * 32 + lane) * 8;
      Bh[ks] = *(const v16bf*)(Bhq + off);
      Bl[ks] = *(const v16bf*)(Blq + off);
    }
    const int j = pblock * 112 + w * 16 + (lane & 15);
    bv       = (j < SP_COLS) ? bias[j] : 0.0f;
    colLocal = w * 16 + (lane & 15);
    rowBase  = (lane >> 4) * 8;
    abase    = (size_t)b * SP_TTILES * SP_KSTEPS * 32 * 8;

    // prologue: stage tile 0 into A-buffer 0 (chunks striped over 7 waves)
#pragma unroll 1
    for (int c = w; c < 2 * SP_KSTEPS; c += 7) {
      const int part = c & 1, ks = c >> 1;
      const unsigned* g = (part ? Alq : Ahq) + abase +
                          ((size_t)ks * 32 + lane) * 8;
      copy32_to_lds(g, &aLDS[0][part][ks][lane][0]);
    }
  }

  // ---- scan-wave persistent state ----
  const int  pglob = pblock * 8 + lane;
  const bool on = (!gemmWave) && (lane < 8) && (pglob < SP_P);
  float ep[6], hid[7], sc = SP_ZERO;
#pragma unroll
  for (int l = 0; l < 6; ++l) ep[l] = on ? epsilon[pglob * 6 + l] : 0.0f;
#pragma unroll
  for (int l = 0; l < 7; ++l) hid[l] = SP_ZERO;
  hid[0] = 0.0f;
  const int  dlen = doc_lens[b];
  const bool e6 = pglob >= 100, e5 = pglob >= 50;
  const int  lp14 = lane * 14;

  for (int it = 0; it <= SP_TTILES; ++it) {
    if (gemmWave) wait_async_copies();   // this wave's tile-`it` chunks landed
    __syncthreads();                     // all waves' chunks landed/visible
    if (gemmWave) {
      if (it + 1 < SP_TTILES) {          // prefetch tile it+1 into buf (it+1)&1
        const int nbuf = (it + 1) & 1;
        const size_t tb = abase + (size_t)(it + 1) * SP_KSTEPS * 32 * 8;
#pragma unroll 1
        for (int c = w; c < 2 * SP_KSTEPS; c += 7) {
          const int part = c & 1, ks = c >> 1;
          const unsigned* g = (part ? Alq : Ahq) + tb +
                              ((size_t)ks * 32 + lane) * 8;
          copy32_to_lds(g, &aLDS[nbuf][part][ks][lane][0]);
        }
      }
      if (it < SP_TTILES) {
        const int buf = it & 1;
        v8f c = {};
        // software-pipelined A operands: load K-step ks+1 while WMMAs of
        // step ks execute (hides the ds_load latency behind 3 WMMAs)
        v16bf ah = *(const v16bf*)(&aLDS[buf][0][0][lane][0]);
        v16bf al = *(const v16bf*)(&aLDS[buf][1][0][lane][0]);
#pragma unroll
        for (int ks = 0; ks < SP_KSTEPS; ++ks) {
          v16bf ah_n = ah, al_n = al;
          if (ks + 1 < SP_KSTEPS) {
            ah_n = *(const v16bf*)(&aLDS[buf][0][ks + 1][lane][0]);
            al_n = *(const v16bf*)(&aLDS[buf][1][ks + 1][lane][0]);
          }
          // bf16x3: hi*hi + hi*lo + lo*hi ~= fp32 product
          c = __builtin_amdgcn_wmma_f32_16x16x32_bf16(false, ah, false, Bh[ks],
                                                      (short)0, c, false, false);
          c = __builtin_amdgcn_wmma_f32_16x16x32_bf16(false, ah, false, Bl[ks],
                                                      (short)0, c, false, false);
          c = __builtin_amdgcn_wmma_f32_16x16x32_bf16(false, al, false, Bh[ks],
                                                      (short)0, c, false, false);
          ah = ah_n;
          al = al_n;
        }
#pragma unroll
        for (int r = 0; r < 8; ++r)
          tmLDS[buf][rowBase + r][colLocal] = c[r] + bv;
      }
    } else if (it >= 1 && on) {
      const int buf = (it - 1) & 1;
      for (int s = 0; s < 16; ++s) {
        const float* row = &tmLDS[buf][s][0];
        float ae[7];
        ae[0] = fmaxf(hid[0], SP_ZERO);
#pragma unroll
        for (int l = 1; l < 7; ++l) ae[l] = fmaxf(hid[l], hid[l - 1] + ep[l - 1]);
        float nh[7];
        nh[0] = fmaxf(0.0f, hid[0] + row[lp14 + 0]);
#pragma unroll
        for (int l = 1; l < 7; ++l)
          nh[l] = fmaxf(ae[l - 1] + row[lp14 + 7 + (l - 1)],
                        hid[l] + row[lp14 + l]);
#pragma unroll
        for (int l = 0; l < 7; ++l) hid[l] = nh[l];
        const float ev = e6 ? hid[6] : (e5 ? hid[5] : hid[4]);
        const int t = (it - 1) * 16 + s;
        if (t < dlen) sc = fmaxf(sc, ev);
      }
    }
  }
  if (on) scores[b * SP_P + pglob] = sc;
}

// ---------------------------------------------------------------------------
__global__ void __launch_bounds__(128) mlp_kernel(
    const float* __restrict__ scores, const float* __restrict__ w0,
    const float* __restrict__ b0, const float* __restrict__ w1,
    const float* __restrict__ b1, float* __restrict__ out) {
  __shared__ float h[100];
  const int b = blockIdx.x;
  const int i = threadIdx.x;
  if (i < 100) {
    float acc = b0[i];
    for (int p = 0; p < SP_P; ++p) acc += scores[b * SP_P + p] * w0[p * 100 + i];
    h[i] = fmaxf(acc, 0.0f);
  }
  __syncthreads();
  if (i < 2) {
    float acc = b1[i];
    for (int q = 0; q < 100; ++q) acc += h[q] * w1[q * 2 + i];
    out[b * 2 + i] = acc;
  }
}

// ---------------------------------------------------------------------------
extern "C" void kernel_launch(void* const* d_in, const int* in_sizes, int n_in,
                              void* d_out, int out_size, void* d_ws,
                              size_t ws_size, hipStream_t stream) {
  const float* emb   = (const float*)d_in[0];
  const float* diags = (const float*)d_in[1];
  const float* bias  = (const float*)d_in[2];
  const float* eps   = (const float*)d_in[3];
  const float* w0    = (const float*)d_in[4];
  const float* b0    = (const float*)d_in[5];
  const float* w1    = (const float*)d_in[6];
  const float* b1    = (const float*)d_in[7];
  const int*   docs  = (const int*)d_in[8];
  const int*   dlen  = (const int*)d_in[9];
  float*       out   = (float*)d_out;

  char* ws = (char*)d_ws;
  const size_t szB = (size_t)SP_CTILES * SP_KSTEPS * 32 * 8 * 4;         // 1.36 MB
  const size_t szA = (size_t)SP_B * SP_TTILES * SP_KSTEPS * 32 * 8 * 4;  // 41.9 MB
  unsigned* Bh = (unsigned*)(ws);
  unsigned* Bl = (unsigned*)(ws + szB);
  size_t offA  = (2 * szB + 255) & ~(size_t)255;
  unsigned* Ah = (unsigned*)(ws + offA);
  unsigned* Al = (unsigned*)(ws + offA + szA);
  float* scores = (float*)(ws + offA + 2 * szA);  // total ~86.6 MB

  prep_b_kernel<<<SP_CTILES, 320, 0, stream>>>(diags, Bh, Bl);
  prep_a_kernel<<<dim3(SP_TTILES, SP_B), 320, 0, stream>>>(emb, docs, Ah, Al);
  sopa_main_kernel<<<dim3(SP_PBLOCKS, SP_B), 256, 0, stream>>>(
      Ah, Al, Bh, Bl, bias, eps, dlen, scores);
  mlp_kernel<<<SP_B, 128, 0, stream>>>(scores, w0, b0, w1, b1, out);
}
